// LSTMAgent_11416023072920
// MI455X (gfx1250) — matrix operand order repaired and checked by reading, not compiled, not run on hardware
//
#include <hip/hip_runtime.h>

// ---------------------------------------------------------------------------
// LSTM agent forward for MI455X (gfx1250, wave32, WMMA bf16 path).
// All GEMM-shaped math -> v_wmma_f32_16x16x32_bf16, f32 accumulate.
// B (weights) pre-packed into fragment order -> 2x global_load_b128 / frag.
// conv1 stages its image in LDS (x = 402MB > L2, im2col re-reads ~4x).
// ---------------------------------------------------------------------------

typedef __attribute__((ext_vector_type(16))) __bf16 bf16x16;
typedef __attribute__((ext_vector_type(8)))  __bf16 bf16x8;
typedef __attribute__((ext_vector_type(4)))  __bf16 bf16x4;
typedef __attribute__((ext_vector_type(8)))  float  f32x8;

#define LOG2PI_F 1.8378770664093453f

__device__ __forceinline__ __bf16 to_bf(float x) { return (__bf16)x; }

__device__ __forceinline__ bf16x8 ld8(const __bf16* p) {
  return *(const bf16x8*)p;   // 16-byte vector load (global_load_b128)
}

// 8 bf16 from an 8-byte-aligned LDS address (2x ds_load_b64)
__device__ __forceinline__ bf16x8 ld8_lds(const __bf16* p) {
  bf16x4 a = *(const bf16x4*)p;
  bf16x4 b = *(const bf16x4*)(p + 4);
  return __builtin_shufflevector(a, b, 0, 1, 2, 3, 4, 5, 6, 7);
}

__device__ __forceinline__ bf16x16 frag_cat(bf16x8 lo, bf16x8 hi) {
  return __builtin_shufflevector(lo, hi, 0, 1, 2, 3, 4, 5, 6, 7,
                                 8, 9, 10, 11, 12, 13, 14, 15);
}

__device__ __forceinline__ f32x8 wmma_bf16(bf16x16 a, bf16x16 b, f32x8 c) {
  return __builtin_amdgcn_wmma_f32_16x16x32_bf16(false, a, false, b, (short)0,
                                                 c, false, false);
}

// A fragment from a row-major row (per-lane row pointer), K-block kb.
__device__ __forceinline__ bf16x16 load_a_row(const __bf16* __restrict__ row, int kb) {
  int half = (threadIdx.x & 31) >> 4;
  const __bf16* p = row + kb + half * 8;
  return frag_cat(ld8(p), ld8(p + 16));
}

// B fragment from packed layout [kblock][ntile][lane][16] -> 32 contiguous bytes.
__device__ __forceinline__ bf16x16 load_b_packed(const __bf16* __restrict__ Bp,
                                                 int kbi, int ntiles, int nt) {
  int lane = threadIdx.x & 31;
  const __bf16* p = Bp + (((size_t)(kbi * ntiles + nt) * 32) + lane) * 16;
  return frag_cat(ld8(p), ld8(p + 8));
}

// Store C tiles with bias + activation (0 none, 1 relu, 2 tanh) as bf16.
template <int NT>
__device__ __forceinline__ void store_act(const f32x8* acc, __bf16* __restrict__ out,
                                          int ldo, int row0, int col0,
                                          const float* __restrict__ bias, int act) {
  const int lane = threadIdx.x & 31;
  const int rbase = row0 + 8 * (lane >> 4);
#pragma unroll
  for (int t = 0; t < NT; ++t) {
    int col = col0 + 16 * t + (lane & 15);
    float bv = bias ? bias[col] : 0.f;
#pragma unroll
    for (int r = 0; r < 8; ++r) {
      float v = acc[t][r] + bv;
      if (act == 1) v = fmaxf(v, 0.f);
      else if (act == 2) v = tanhf(v);
      out[(size_t)(rbase + r) * ldo + col] = to_bf(v);
    }
  }
}

// ---------------------------------------------------------------------------
// Weight prep: pack every B matrix [KxN] into WMMA fragment order:
// dst[(((kbi*ntiles+nt)*32)+lane)*16+j] = B[kbi*32 + 8*(lane>>4) + (j&7) + 16*(j>>3)][nt*16 + (lane&15)]
// ---------------------------------------------------------------------------
template <typename F>
__device__ __forceinline__ void pack_b_loop(__bf16* __restrict__ dst, int K, int N,
                                            int i0, int st, F&& src) {
  int ntiles = N >> 4;
  int total = K * N;
  for (int i = i0; i < total; i += st) {
    int j = i & 15;
    int l = (i >> 4) & 31;
    int rest = i >> 9;
    int nt = rest % ntiles;
    int kbi = rest / ntiles;
    int k = kbi * 32 + 8 * (l >> 4) + (j & 7) + 16 * (j >> 3);
    int n = nt * 16 + (l & 15);
    dst[i] = src(k, n);
  }
}

__global__ void prep_weights_k(
    const float* __restrict__ c1w, const float* __restrict__ c2w,
    const float* __restrict__ c3w, const float* __restrict__ linw,
    const float* __restrict__ wih, const float* __restrict__ whh,
    const float* __restrict__ bih, const float* __restrict__ bhh,
    const float* __restrict__ aw1, const float* __restrict__ aw2,
    const float* __restrict__ cw1, const float* __restrict__ cw2,
    __bf16* p_c1w, __bf16* p_c2w, __bf16* p_c3w, __bf16* p_linw,
    __bf16* p_wih, __bf16* p_whh, float* d_biasz,
    __bf16* p_aw1, __bf16* p_aw2, __bf16* p_cw1, __bf16* p_cw2) {
  int i0 = blockIdx.x * blockDim.x + threadIdx.x;
  int st = gridDim.x * blockDim.x;
  pack_b_loop(p_c1w, 192, 32, i0, st,
              [&](int k, int n) { return to_bf(c1w[k * 32 + n]); });
  pack_b_loop(p_c2w, 512, 64, i0, st,
              [&](int k, int n) { return to_bf(c2w[k * 64 + n]); });
  pack_b_loop(p_c3w, 576, 64, i0, st,
              [&](int k, int n) { return to_bf(c3w[k * 64 + n]); });
  pack_b_loop(p_linw, 1024, 512, i0, st,
              [&](int k, int n) { return to_bf(linw[(size_t)k * 512 + n]); });
  pack_b_loop(p_wih, 544, 2048, i0, st,
              [&](int k, int n) -> __bf16 {
                return (k < 518) ? to_bf(wih[(size_t)n * 518 + k]) : (__bf16)0.f;
              });
  pack_b_loop(p_whh, 512, 2048, i0, st,
              [&](int k, int n) { return to_bf(whh[(size_t)n * 512 + k]); });
  pack_b_loop(p_aw1, 512, 256, i0, st,
              [&](int k, int n) { return to_bf(aw1[k * 256 + n]); });
  pack_b_loop(p_aw2, 256, 256, i0, st,
              [&](int k, int n) { return to_bf(aw2[k * 256 + n]); });
  pack_b_loop(p_cw1, 512, 256, i0, st,
              [&](int k, int n) { return to_bf(cw1[k * 256 + n]); });
  pack_b_loop(p_cw2, 256, 256, i0, st,
              [&](int k, int n) { return to_bf(cw2[k * 256 + n]); });
  for (int i = i0; i < 2048; i += st) d_biasz[i] = bih[i] + bhh[i];
}

// ---------------------------------------------------------------------------
// conv1: M=8192*225, N=32, K=192 (8x8x3), stride 4, fp32 in -> bf16 relu out.
// One block per image; image staged once in LDS as bf16 (24 KB).
// Kernel row = 24 elems = 3 aligned 8-chunks -> ds_load_b64 pairs.
// ---------------------------------------------------------------------------
__global__ void conv1_k(const float* __restrict__ x, const __bf16* __restrict__ W,
                        const float* __restrict__ bias, __bf16* __restrict__ out) {
  __shared__ __align__(16) __bf16 simg[12288];   // 64*64*3
  const int img = blockIdx.x;
  const int tid = threadIdx.x;
  const float* src = x + (size_t)img * 12288;
  for (int i = tid; i < 3072; i += 160) {
    float4 v = ((const float4*)src)[i];
    bf16x4 b = {to_bf(v.x), to_bf(v.y), to_bf(v.z), to_bf(v.w)};
    *(bf16x4*)&simg[i * 4] = b;
  }
  __syncthreads();
  const int lane = tid & 31, wave = tid >> 5, half = lane >> 4;
  for (int mt = 0; mt < 3; ++mt) {
    int mtile = wave * 3 + mt;                 // 5 waves x 3 = 15 tiles (240 rows)
    int rl = mtile * 16 + (lane & 15);
    int rc = rl < 225 ? rl : 224;              // clamp; masked on store
    int oy = rc / 15, ox = rc - oy * 15;
    const __bf16* sp = simg + oy * 768 + ox * 12;   // (oy*4)*64*3 + (ox*4)*3
    f32x8 acc[2] = {};
#pragma unroll
    for (int kb = 0; kb < 192; kb += 32) {
      int k0 = kb + half * 8;
      int k1 = k0 + 16;
      int ky0 = k0 / 24, r0 = k0 - ky0 * 24;   // r0 in {0,8,16}
      int ky1 = k1 / 24, r1 = k1 - ky1 * 24;
      bf16x16 a = frag_cat(ld8_lds(sp + ky0 * 192 + r0),
                           ld8_lds(sp + ky1 * 192 + r1));
#pragma unroll
      for (int t = 0; t < 2; ++t)
        acc[t] = wmma_bf16(a, load_b_packed(W, kb >> 5, 2, t), acc[t]);
    }
    int rbase = mtile * 16 + 8 * half;
#pragma unroll
    for (int t = 0; t < 2; ++t) {
      int col = 16 * t + (lane & 15);
      float bv = bias[col];
#pragma unroll
      for (int r = 0; r < 8; ++r) {
        int row = rbase + r;
        if (row < 225) {
          float v = fmaxf(acc[t][r] + bv, 0.f);
          out[((size_t)img * 225 + row) * 32 + col] = to_bf(v);
        }
      }
    }
  }
}

// conv2: M=8192*36, N=64, K=512 (4x4x32), stride 2.  Source fits L2; direct gather.
__global__ void conv2_k(const __bf16* __restrict__ in, const __bf16* __restrict__ W,
                        const float* __restrict__ bias, __bf16* __restrict__ out) {
  int wid = (blockIdx.x * blockDim.x + threadIdx.x) >> 5;
  int lane = threadIdx.x & 31;
  int half = lane >> 4;
  int arow = wid * 16 + (lane & 15);
  int img = arow / 36; int p = arow - img * 36;
  int oy = p / 6;      int ox = p - oy * 6;
  const __bf16* src = in + ((size_t)img * 225 + (oy * 2) * 15 + ox * 2) * 32;
  f32x8 acc[4] = {};
#pragma unroll
  for (int kb = 0; kb < 512; kb += 32) {
    int k0 = kb + half * 8, k1 = kb + 16 + half * 8;
    int p0 = k0 >> 5, p1 = k1 >> 5;
    bf16x16 a = frag_cat(
        ld8(src + ((p0 >> 2) * 15 + (p0 & 3)) * 32 + (k0 & 31)),
        ld8(src + ((p1 >> 2) * 15 + (p1 & 3)) * 32 + (k1 & 31)));
#pragma unroll
    for (int t = 0; t < 4; ++t)
      acc[t] = wmma_bf16(a, load_b_packed(W, kb >> 5, 4, t), acc[t]);
  }
  store_act<4>(acc, out, 64, wid * 16, 0, bias, 1);
}

// conv3: M=8192*16, N=64, K=576 (3x3x64), stride 1.
__global__ void conv3_k(const __bf16* __restrict__ in, const __bf16* __restrict__ W,
                        const float* __restrict__ bias, __bf16* __restrict__ out) {
  int wid = (blockIdx.x * blockDim.x + threadIdx.x) >> 5;
  int lane = threadIdx.x & 31;
  int half = lane >> 4;
  int arow = wid * 16 + (lane & 15);
  int img = arow >> 4; int p = arow & 15;
  int oy = p >> 2;     int ox = p & 3;
  const __bf16* src = in + ((size_t)img * 36 + oy * 6 + ox) * 64;
  f32x8 acc[4] = {};
#pragma unroll
  for (int kb = 0; kb < 576; kb += 32) {
    int k0 = kb + half * 8, k1 = kb + 16 + half * 8;
    int p0 = k0 >> 6, p1 = k1 >> 6;
    int ky0 = p0 / 3, kx0 = p0 - ky0 * 3;
    int ky1 = p1 / 3, kx1 = p1 - ky1 * 3;
    bf16x16 a = frag_cat(ld8(src + (ky0 * 6 + kx0) * 64 + (k0 & 63)),
                         ld8(src + (ky1 * 6 + kx1) * 64 + (k1 & 63)));
#pragma unroll
    for (int t = 0; t < 4; ++t)
      acc[t] = wmma_bf16(a, load_b_packed(W, kb >> 5, 4, t), acc[t]);
  }
  store_act<4>(acc, out, 64, wid * 16, 0, bias, 1);
}

// linear: M=8192, N=512, K=1024, relu, writes into xs rows (ld 544, cols 0..511)
__global__ void lin_k(const __bf16* __restrict__ A, const __bf16* __restrict__ W,
                      const float* __restrict__ bias, __bf16* __restrict__ xs) {
  int wid = (blockIdx.x * blockDim.x + threadIdx.x) >> 5;
  int lane = threadIdx.x & 31;
  int mtile = wid >> 3, ng = wid & 7;
  const __bf16* Ar = A + (size_t)(mtile * 16 + (lane & 15)) * 1024;
  f32x8 acc[4] = {};
  for (int kb = 0; kb < 1024; kb += 32) {
    __builtin_prefetch(Ar + kb + 256, 0, 0);
    bf16x16 a = load_a_row(Ar, kb);
#pragma unroll
    for (int t = 0; t < 4; ++t)
      acc[t] = wmma_bf16(a, load_b_packed(W, kb >> 5, 32, ng * 4 + t), acc[t]);
  }
  store_act<4>(acc, xs, 544, mtile * 16, ng * 64, bias, 1);
}

// Fill xs cols 512..543: action(3), reward, contact, time_till_end, zero pad
__global__ void xs_extras_k(const float* __restrict__ la, const float* __restrict__ lr,
                            const float* __restrict__ ct, const float* __restrict__ tt,
                            __bf16* __restrict__ xs) {
  int gm = blockIdx.x * blockDim.x + threadIdx.x;
  if (gm >= 8192) return;
  __bf16* row = xs + (size_t)gm * 544 + 512;
  row[0] = to_bf(la[gm * 3 + 0]);
  row[1] = to_bf(la[gm * 3 + 1]);
  row[2] = to_bf(la[gm * 3 + 2]);
  row[3] = to_bf(lr[gm]);
  row[4] = to_bf(ct[gm]);
  row[5] = to_bf(tt[gm]);
#pragma unroll
  for (int j = 6; j < 32; ++j) row[j] = (__bf16)0.f;
}

// ---------------------------------------------------------------------------
// LSTM scan
// ---------------------------------------------------------------------------
__global__ void lstm_init_k(const float* __restrict__ h0, const float* __restrict__ c0,
                            float* __restrict__ h, float* __restrict__ c) {
  int i = blockIdx.x * blockDim.x + threadIdx.x;
  if (i < 65536) { h[i] = h0[i]; c[i] = c0[i]; }
}

__global__ void lstm_mask_k(const int* __restrict__ done_t, float* __restrict__ h,
                            float* __restrict__ c, __bf16* __restrict__ hm) {
  int i = blockIdx.x * blockDim.x + threadIdx.x;
  if (i >= 65536) return;
  float m = done_t[i >> 9] ? 0.f : 1.f;
  float hv = h[i] * m;
  h[i] = hv; c[i] *= m;
  hm[i] = to_bf(hv);
}

// z[128][2048] = xs_t @ wihT + h @ whhT + (b_ih + b_hh), f32 out
__global__ void lstm_gemm_k(const __bf16* __restrict__ xs_t, const __bf16* __restrict__ hm,
                            const __bf16* __restrict__ wihP, const __bf16* __restrict__ whhP,
                            const float* __restrict__ biasz, float* __restrict__ z) {
  int wid = (blockIdx.x * blockDim.x + threadIdx.x) >> 5;
  int lane = threadIdx.x & 31;
  int mtile = wid >> 5;     // 8 mtiles
  int ng = wid & 31;        // 32 groups of 4 ntiles (64 cols)
  const __bf16* a1 = xs_t + (size_t)(mtile * 16 + (lane & 15)) * 544;
  const __bf16* a2 = hm + (size_t)(mtile * 16 + (lane & 15)) * 512;
  f32x8 acc[4] = {};
  for (int kb = 0; kb < 544; kb += 32) {
    bf16x16 a = load_a_row(a1, kb);
#pragma unroll
    for (int t = 0; t < 4; ++t)
      acc[t] = wmma_bf16(a, load_b_packed(wihP, kb >> 5, 128, ng * 4 + t), acc[t]);
  }
  for (int kb = 0; kb < 512; kb += 32) {
    bf16x16 a = load_a_row(a2, kb);
#pragma unroll
    for (int t = 0; t < 4; ++t)
      acc[t] = wmma_bf16(a, load_b_packed(whhP, kb >> 5, 128, ng * 4 + t), acc[t]);
  }
  int rbase = mtile * 16 + 8 * (lane >> 4);
#pragma unroll
  for (int t = 0; t < 4; ++t) {
    int col = ng * 64 + 16 * t + (lane & 15);
    float bv = biasz[col];
#pragma unroll
    for (int r = 0; r < 8; ++r)
      z[(size_t)(rbase + r) * 2048 + col] = acc[t][r] + bv;
  }
}

__global__ void lstm_gate_k(const float* __restrict__ z, float* __restrict__ h,
                            float* __restrict__ c, __bf16* __restrict__ hid_t) {
  int i = blockIdx.x * blockDim.x + threadIdx.x;
  if (i >= 65536) return;
  int b = i >> 9, j = i & 511;
  const float* zr = z + (size_t)b * 2048;
  float zi = zr[j], zf = zr[512 + j], zg = zr[1024 + j], zo = zr[1536 + j];
  float si = 1.f / (1.f + __expf(-zi));
  float sf = 1.f / (1.f + __expf(-zf));
  float so = 1.f / (1.f + __expf(-zo));
  float cn = sf * c[i] + si * tanhf(zg);
  float hn = so * tanhf(cn);
  c[i] = cn; h[i] = hn;
  hid_t[i] = to_bf(hn);
}

// ---------------------------------------------------------------------------
// MLP heads: M=8192, N=256, tanh epilogue
// ---------------------------------------------------------------------------
__global__ void mlp_gemm_k(const __bf16* __restrict__ A, int K,
                           const __bf16* __restrict__ B, const float* __restrict__ bias,
                           __bf16* __restrict__ out) {
  int wid = (blockIdx.x * blockDim.x + threadIdx.x) >> 5;
  int lane = threadIdx.x & 31;
  int mtile = wid >> 2, ng = wid & 3;
  const __bf16* Ar = A + (size_t)(mtile * 16 + (lane & 15)) * K;
  f32x8 acc[4] = {};
  for (int kb = 0; kb < K; kb += 32) {
    __builtin_prefetch(Ar + kb + 128, 0, 0);
    bf16x16 a = load_a_row(Ar, kb);
#pragma unroll
    for (int t = 0; t < 4; ++t)
      acc[t] = wmma_bf16(a, load_b_packed(B, kb >> 5, 16, ng * 4 + t), acc[t]);
  }
  store_act<4>(acc, out, 256, mtile * 16, ng * 64, bias, 2);
}

// Final tiny layers + logprob/entropy/value
__global__ void head_k(const __bf16* __restrict__ a2, const float* __restrict__ aw3,
                       const float* __restrict__ ab3, const float* __restrict__ logstd,
                       const __bf16* __restrict__ c2, const float* __restrict__ cw3,
                       const float* __restrict__ cb3, const float* __restrict__ action,
                       float* __restrict__ out) {
  int gm = blockIdx.x * blockDim.x + threadIdx.x;
  if (gm >= 8192) return;
  float m0 = ab3[0], m1 = ab3[1], m2 = ab3[2], v = cb3[0];
  const __bf16* ar = a2 + (size_t)gm * 256;
  const __bf16* cr = c2 + (size_t)gm * 256;
  for (int k = 0; k < 256; ++k) {
    float av = (float)ar[k];
    m0 += av * aw3[k * 3 + 0];
    m1 += av * aw3[k * 3 + 1];
    m2 += av * aw3[k * 3 + 2];
    v += (float)cr[k] * cw3[k];
  }
  float mean[3] = {m0, m1, m2};
  float lp = 0.f, ent = 0.f;
#pragma unroll
  for (int j = 0; j < 3; ++j) {
    float ls = logstd[j];
    float iv = __expf(-2.f * ls);
    float d = action[gm * 3 + j] - mean[j];
    lp += -0.5f * d * d * iv - ls - 0.5f * LOG2PI_F;
    ent += 0.5f + 0.5f * LOG2PI_F + ls;
  }
  out[gm] = lp;
  out[8192 + gm] = ent;
  out[16384 + gm] = v;
}

__global__ void write_states_k(const float* __restrict__ h, const float* __restrict__ c,
                               float* __restrict__ out) {
  int i = blockIdx.x * blockDim.x + threadIdx.x;
  if (i < 65536) {
    out[24576 + i] = h[i];
    out[24576 + 65536 + i] = c[i];
  }
}

// ---------------------------------------------------------------------------
// Host launch
// ---------------------------------------------------------------------------
extern "C" void kernel_launch(void* const* d_in, const int* in_sizes, int n_in,
                              void* d_out, int out_size, void* d_ws, size_t ws_size,
                              hipStream_t stream) {
  (void)in_sizes; (void)n_in; (void)out_size; (void)ws_size;
  const float* x        = (const float*)d_in[0];
  const int*   done     = (const int*)d_in[1];
  const float* last_act = (const float*)d_in[2];
  const float* last_rew = (const float*)d_in[3];
  const float* contact  = (const float*)d_in[4];
  const float* ttend    = (const float*)d_in[5];
  const float* action   = (const float*)d_in[6];
  const float* h0       = (const float*)d_in[7];
  const float* c0       = (const float*)d_in[8];
  const float* conv1_w  = (const float*)d_in[9];
  const float* conv1_b  = (const float*)d_in[10];
  const float* conv2_w  = (const float*)d_in[11];
  const float* conv2_b  = (const float*)d_in[12];
  const float* conv3_w  = (const float*)d_in[13];
  const float* conv3_b  = (const float*)d_in[14];
  const float* lin_w    = (const float*)d_in[15];
  const float* lin_b    = (const float*)d_in[16];
  const float* w_ih     = (const float*)d_in[17];
  const float* w_hh     = (const float*)d_in[18];
  const float* b_ih     = (const float*)d_in[19];
  const float* b_hh     = (const float*)d_in[20];
  const float* cr_w1    = (const float*)d_in[21];
  const float* cr_b1    = (const float*)d_in[22];
  const float* cr_w2    = (const float*)d_in[23];
  const float* cr_b2    = (const float*)d_in[24];
  const float* cr_w3    = (const float*)d_in[25];
  const float* cr_b3    = (const float*)d_in[26];
  const float* ac_w1    = (const float*)d_in[27];
  const float* ac_b1    = (const float*)d_in[28];
  const float* ac_w2    = (const float*)d_in[29];
  const float* ac_b2    = (const float*)d_in[30];
  const float* ac_w3    = (const float*)d_in[31];
  const float* ac_b3    = (const float*)d_in[32];
  const float* a_logstd = (const float*)d_in[33];
  float* out = (float*)d_out;

  char* ws = (char*)d_ws;
  size_t off = 0;
  auto alloc = [&](size_t bytes) -> char* {
    char* p = ws + off;
    off += (bytes + 255) & ~(size_t)255;
    return p;
  };
  // Persistent packed weights (bf16) + state
  __bf16* p_c1w  = (__bf16*)alloc(6144 * 2);
  __bf16* p_c2w  = (__bf16*)alloc(32768 * 2);
  __bf16* p_c3w  = (__bf16*)alloc(36864 * 2);
  __bf16* p_linw = (__bf16*)alloc(524288 * 2);
  __bf16* p_wih  = (__bf16*)alloc((size_t)544 * 2048 * 2);
  __bf16* p_whh  = (__bf16*)alloc((size_t)512 * 2048 * 2);
  float*  d_bz   = (float*)alloc(2048 * 4);
  __bf16* p_aw1  = (__bf16*)alloc(131072 * 2);
  __bf16* p_aw2  = (__bf16*)alloc(65536 * 2);
  __bf16* p_cw1  = (__bf16*)alloc(131072 * 2);
  __bf16* p_cw2  = (__bf16*)alloc(65536 * 2);
  float*  d_h    = (float*)alloc(65536 * 4);
  float*  d_c    = (float*)alloc(65536 * 4);
  __bf16* d_hm   = (__bf16*)alloc(65536 * 2);
  float*  d_z    = (float*)alloc((size_t)262144 * 4);
  // conv2 output (live across conv2->conv3 only)
  __bf16* d_c2o  = (__bf16*)alloc((size_t)18874368 * 2);
  // Big aliased region: conv1 output (118 MB) reused after conv2 completes
  char* bigA = alloc((size_t)117964800);
  __bf16* d_c1o  = (__bf16*)bigA;                              // 8192*225*32
  __bf16* d_c3o  = (__bf16*)bigA;                              // 8192*16*64 (conv1 dead)
  __bf16* d_xs   = (__bf16*)(bigA + (size_t)20 * 1024 * 1024); // 8192*544
  __bf16* d_hid  = (__bf16*)(bigA + (size_t)32 * 1024 * 1024); // 8192*512
  __bf16* d_a1   = (__bf16*)(bigA + (size_t)48 * 1024 * 1024); // 8192*256
  __bf16* d_a2   = (__bf16*)(bigA + (size_t)56 * 1024 * 1024);
  __bf16* d_cc1  = (__bf16*)(bigA + (size_t)64 * 1024 * 1024);
  __bf16* d_cc2  = (__bf16*)(bigA + (size_t)72 * 1024 * 1024);

  // 1. weight prep (fragment-order packing)
  prep_weights_k<<<2048, 256, 0, stream>>>(
      conv1_w, conv2_w, conv3_w, lin_w, w_ih, w_hh, b_ih, b_hh,
      ac_w1, ac_w2, cr_w1, cr_w2,
      p_c1w, p_c2w, p_c3w, p_linw, p_wih, p_whh, d_bz,
      p_aw1, p_aw2, p_cw1, p_cw2);

  // 2. CNN encoder (implicit-GEMM WMMA; conv1 LDS-staged per image)
  conv1_k<<<8192, 160, 0, stream>>>(x, p_c1w, conv1_b, d_c1o);
  conv2_k<<<4608, 128, 0, stream>>>(d_c1o, p_c2w, conv2_b, d_c2o); // 18432 waves
  conv3_k<<<2048, 128, 0, stream>>>(d_c2o, p_c3w, conv3_b, d_c3o); // 8192 waves
  lin_k<<<1024, 128, 0, stream>>>(d_c3o, p_linw, lin_b, d_xs);     // 4096 waves
  xs_extras_k<<<64, 128, 0, stream>>>(last_act, last_rew, contact, ttend, d_xs);

  // 3. LSTM scan (64 sequential steps)
  lstm_init_k<<<512, 128, 0, stream>>>(h0, c0, d_h, d_c);
  for (int t = 0; t < 64; ++t) {
    lstm_mask_k<<<512, 128, 0, stream>>>(done + t * 128, d_h, d_c, d_hm);
    lstm_gemm_k<<<64, 128, 0, stream>>>(d_xs + (size_t)t * 128 * 544, d_hm,
                                        p_wih, p_whh, d_bz, d_z);
    lstm_gate_k<<<512, 128, 0, stream>>>(d_z, d_h, d_c,
                                         d_hid + (size_t)t * 65536);
  }

  // 4. actor / critic heads
  mlp_gemm_k<<<512, 128, 0, stream>>>(d_hid, 512, p_aw1, ac_b1, d_a1);
  mlp_gemm_k<<<512, 128, 0, stream>>>(d_a1, 256, p_aw2, ac_b2, d_a2);
  mlp_gemm_k<<<512, 128, 0, stream>>>(d_hid, 512, p_cw1, cr_b1, d_cc1);
  mlp_gemm_k<<<512, 128, 0, stream>>>(d_cc1, 256, p_cw2, cr_b2, d_cc2);
  head_k<<<64, 128, 0, stream>>>(d_a2, ac_w3, ac_b3, a_logstd,
                                 d_cc2, cr_w3, cr_b3, action, out);
  write_states_k<<<512, 128, 0, stream>>>(d_h, d_c, out);
}
